// GCNEncoder_7198365188144
// MI455X (gfx1250) — matrix operand order repaired and checked by reading, not compiled, not run on hardware
//
#include <hip/hip_runtime.h>
#include <hip/hip_bf16.h>

typedef __attribute__((ext_vector_type(16))) __bf16 v16bf;
typedef __attribute__((ext_vector_type(8)))  float  v8f;

#define NNODES 100000
#define NEDGES 1600000
#define INC    128
#define HIDC   128
#define EMBC   64

union Frag {                 // 32 bytes: WMMA 16-bit operand as 8 packed dwords
    v16bf bf;
    uint4 q[2];
    unsigned u[8];
};

__device__ __forceinline__ __bf16 bf_hi(float v) { return (__bf16)v; }
__device__ __forceinline__ __bf16 bf_lo(float v, __bf16 hi) {
    return (__bf16)(v - (float)hi);
}
__device__ __forceinline__ unsigned pack_bf16(__bf16 a, __bf16 b) {
    union { __bf16 h[2]; unsigned u; } t;
    t.h[0] = a; t.h[1] = b;
    return t.u;
}

// ---------------------------------------------------------------------------
// Degree / normalization
// ---------------------------------------------------------------------------
__global__ void k_deg_init(float* __restrict__ deg) {
    int i = blockIdx.x * blockDim.x + threadIdx.x;
    if (i < NNODES) deg[i] = 1.0f;              // self-loop
}

__global__ void k_deg_count(const long long* __restrict__ col, float* __restrict__ deg) {
    int e = blockIdx.x * blockDim.x + threadIdx.x;
    if (e < NEDGES) unsafeAtomicAdd(&deg[(int)col[e]], 1.0f);
}

__global__ void k_dinv(float* __restrict__ deg) {
    int i = blockIdx.x * blockDim.x + threadIdx.x;
    if (i < NNODES) deg[i] = rsqrtf(deg[i]);
}

// ---------------------------------------------------------------------------
// Pre-split weights: W[K][N] fp32 -> transposed pair-packed bf16 hi/lo,
// hiT/loT layout [N][K/2] dwords; dword p of column n = bf16(W[2p][n]) |
// bf16(W[2p+1][n])<<16. This is exactly the B-fragment register image.
// ---------------------------------------------------------------------------
__global__ void k_splitW(const float* __restrict__ W, int K, int N,
                         unsigned* __restrict__ hiT, unsigned* __restrict__ loT) {
    int idx = blockIdx.x * blockDim.x + threadIdx.x;
    int kd = K >> 1;
    if (idx >= N * kd) return;
    int n = idx / kd, p = idx % kd;
    float w0 = W[(size_t)(2 * p) * N + n];
    float w1 = W[(size_t)(2 * p + 1) * N + n];
    __bf16 h0 = bf_hi(w0), h1 = bf_hi(w1);
    hiT[idx] = pack_bf16(h0, h1);
    loT[idx] = pack_bf16(bf_lo(w0, h0), bf_lo(w1, h1));
}

// ---------------------------------------------------------------------------
// GEMM1: xw = x @ W1   [100000x128] = [100000x128][128x128]
// Block = 256 threads = 8 waves; block owns 16 rows, wave owns one 16x16
// n-tile. A-tile split to bf16 hi/lo once in LDS (pair-packed, stride 68
// dwords -> 16B-aligned ds_load_b128, conflict-free). bf16x3 accumulation.
// ---------------------------------------------------------------------------
__launch_bounds__(256)
__global__ void k_gemm1(const float* __restrict__ x,
                        const unsigned* __restrict__ WhiT, const unsigned* __restrict__ WloT,
                        float* __restrict__ xw) {
    __shared__ unsigned xhi[16 * 68];
    __shared__ unsigned xlo[16 * 68];
    const int mblk = blockIdx.x;                // 0..6249
    const int t    = threadIdx.x;

    const float2* xsrc = (const float2*)(x + (size_t)mblk * 16 * INC);
    for (int idx = t; idx < 16 * 64; idx += 256) {
        int r = idx >> 6, c2 = idx & 63;
        float2 v = xsrc[idx];
        __bf16 h0 = bf_hi(v.x), h1 = bf_hi(v.y);
        xhi[r * 68 + c2] = pack_bf16(h0, h1);
        xlo[r * 68 + c2] = pack_bf16(bf_lo(v.x, h0), bf_lo(v.y, h1));
    }
    __syncthreads();

    const int wave    = t >> 5;                 // n-tile id 0..7
    const int lane    = t & 31;
    const int n0      = wave * 16;
    const int mn      = lane & 15;              // row (A) / col (B,D) within tile
    const int halfsel = lane >> 4;

    v8f acc = {};
    for (int ks = 0; ks < 4; ++ks) {            // K = 128 in chunks of 32
        Frag ah, al, bh, bl;
        // A 16x32: dwords [kbase/2 + (v<4?0:8) + halfsel*4 + (v&3)]
        int abase = mn * 68 + ks * 16 + halfsel * 4;
        ah.q[0] = *(const uint4*)&xhi[abase];
        ah.q[1] = *(const uint4*)&xhi[abase + 8];
        al.q[0] = *(const uint4*)&xlo[abase];
        al.q[1] = *(const uint4*)&xlo[abase + 8];
        // B 32x16: column n0+mn, dwords [kbase/2 + (v<4?0:8) + halfsel*4 + (v&3)]
        int bbase = (n0 + mn) * (HIDC / 2) + ks * 16 + halfsel * 4;
        bh.q[0] = *(const uint4*)&WhiT[bbase];
        bh.q[1] = *(const uint4*)&WhiT[bbase + 8];
        bl.q[0] = *(const uint4*)&WloT[bbase];
        bl.q[1] = *(const uint4*)&WloT[bbase + 8];

        acc = __builtin_amdgcn_wmma_f32_16x16x32_bf16(false, ah.bf, false, bh.bf, (short)0, acc, false, false);
        acc = __builtin_amdgcn_wmma_f32_16x16x32_bf16(false, ah.bf, false, bl.bf, (short)0, acc, false, false);
        acc = __builtin_amdgcn_wmma_f32_16x16x32_bf16(false, al.bf, false, bh.bf, (short)0, acc, false, false);
    }
#pragma unroll
    for (int j = 0; j < 8; ++j) {               // D: VGPR j -> row j (+8 upper half)
        int mm = mblk * 16 + j + (halfsel ? 8 : 0);
        xw[(size_t)mm * HIDC + n0 + mn] = acc[j];
    }
}

// ---------------------------------------------------------------------------
// GEMM2: xw2 = relu(agg1 + b1) @ W2   [100000x64] = [100000x128][128x64]
// Block owns 32 rows; 8 waves -> 2 m-subtiles x 4 n-tiles. Bias+ReLU fused
// into the (single) bf16 split during LDS staging.
// ---------------------------------------------------------------------------
__launch_bounds__(256)
__global__ void k_gemm2(const float* __restrict__ agg, const float* __restrict__ b1,
                        const unsigned* __restrict__ WhiT, const unsigned* __restrict__ WloT,
                        float* __restrict__ xw2) {
    __shared__ unsigned ahi[32 * 68];
    __shared__ unsigned alo[32 * 68];
    const int mblk = blockIdx.x;                // 0..3124
    const int t    = threadIdx.x;

    const float2* asrc = (const float2*)(agg + (size_t)mblk * 32 * HIDC);
    for (int idx = t; idx < 32 * 64; idx += 256) {
        int r = idx >> 6, c2 = idx & 63;
        float2 v = asrc[idx];
        float v0 = v.x + b1[2 * c2];
        float v1 = v.y + b1[2 * c2 + 1];
        v0 = v0 > 0.0f ? v0 : 0.0f;             // ReLU
        v1 = v1 > 0.0f ? v1 : 0.0f;
        __bf16 h0 = bf_hi(v0), h1 = bf_hi(v1);
        ahi[r * 68 + c2] = pack_bf16(h0, h1);
        alo[r * 68 + c2] = pack_bf16(bf_lo(v0, h0), bf_lo(v1, h1));
    }
    __syncthreads();

    const int wave    = t >> 5;
    const int msub    = wave >> 2;              // 0..1
    const int ntile   = wave & 3;               // 0..3
    const int lane    = t & 31;
    const int n0      = ntile * 16;
    const int mn      = lane & 15;
    const int halfsel = lane >> 4;

    v8f acc = {};
    for (int ks = 0; ks < 4; ++ks) {
        Frag ah, al, bh, bl;
        int abase = (msub * 16 + mn) * 68 + ks * 16 + halfsel * 4;
        ah.q[0] = *(const uint4*)&ahi[abase];
        ah.q[1] = *(const uint4*)&ahi[abase + 8];
        al.q[0] = *(const uint4*)&alo[abase];
        al.q[1] = *(const uint4*)&alo[abase + 8];
        int bbase = (n0 + mn) * (HIDC / 2) + ks * 16 + halfsel * 4;   // K=128 -> 64 dwords/col
        bh.q[0] = *(const uint4*)&WhiT[bbase];
        bh.q[1] = *(const uint4*)&WhiT[bbase + 8];
        bl.q[0] = *(const uint4*)&WloT[bbase];
        bl.q[1] = *(const uint4*)&WloT[bbase + 8];

        acc = __builtin_amdgcn_wmma_f32_16x16x32_bf16(false, ah.bf, false, bh.bf, (short)0, acc, false, false);
        acc = __builtin_amdgcn_wmma_f32_16x16x32_bf16(false, ah.bf, false, bl.bf, (short)0, acc, false, false);
        acc = __builtin_amdgcn_wmma_f32_16x16x32_bf16(false, al.bf, false, bh.bf, (short)0, acc, false, false);
    }
#pragma unroll
    for (int j = 0; j < 8; ++j) {
        int mm = mblk * 32 + msub * 16 + j + (halfsel ? 8 : 0);
        xw2[(size_t)mm * EMBC + n0 + mn] = acc[j];
    }
}

// ---------------------------------------------------------------------------
// Self-loop initializers
// ---------------------------------------------------------------------------
__global__ void k_selfinit1(const float* __restrict__ xw, const float* __restrict__ dinv,
                            float* __restrict__ agg) {
    size_t i = (size_t)blockIdx.x * blockDim.x + threadIdx.x;
    if (i < (size_t)NNODES * HIDC) {
        float d = dinv[i >> 7];
        agg[i] = d * d * xw[i];
    }
}

__global__ void k_selfinit2(const float* __restrict__ xw2, const float* __restrict__ dinv,
                            const float* __restrict__ b2, float* __restrict__ out) {
    size_t i = (size_t)blockIdx.x * blockDim.x + threadIdx.x;
    if (i < (size_t)NNODES * EMBC) {
        float d = dinv[i >> 6];
        out[i] = d * d * xw2[i] + b2[i & 63];
    }
}

// ---------------------------------------------------------------------------
// Edge scatter: one wave per edge; coalesced row gather, hw fp32 atomics.
// Working set (xw + agg + edges ~ 128 MB) is L2-resident on MI455X (192 MB).
// ---------------------------------------------------------------------------
__launch_bounds__(256)
__global__ void k_edge1(const long long* __restrict__ rows, const long long* __restrict__ cols,
                        const float* __restrict__ dinv, const float* __restrict__ xw,
                        float* __restrict__ agg) {
    int wid  = (int)(((size_t)blockIdx.x * blockDim.x + threadIdx.x) >> 5);
    int lane = threadIdx.x & 31;
    if (wid >= NEDGES) return;
    int r = (int)rows[wid], c = (int)cols[wid];
    float norm = dinv[r] * dinv[c];
    const float* src = xw + (size_t)r * HIDC;
    float* dst = agg + (size_t)c * HIDC;
#pragma unroll
    for (int i = 0; i < 4; ++i) {
        int ch = lane + i * 32;
        unsafeAtomicAdd(&dst[ch], norm * src[ch]);
    }
}

__launch_bounds__(256)
__global__ void k_edge2(const long long* __restrict__ rows, const long long* __restrict__ cols,
                        const float* __restrict__ dinv, const float* __restrict__ xw2,
                        float* __restrict__ out) {
    int wid  = (int)(((size_t)blockIdx.x * blockDim.x + threadIdx.x) >> 5);
    int lane = threadIdx.x & 31;
    if (wid >= NEDGES) return;
    int r = (int)rows[wid], c = (int)cols[wid];
    float norm = dinv[r] * dinv[c];
    const float* src = xw2 + (size_t)r * EMBC;
    float* dst = out + (size_t)c * EMBC;
#pragma unroll
    for (int i = 0; i < 2; ++i) {
        int ch = lane + i * 32;
        unsafeAtomicAdd(&dst[ch], norm * src[ch]);
    }
}

// ---------------------------------------------------------------------------
// Launcher
// ---------------------------------------------------------------------------
extern "C" void kernel_launch(void* const* d_in, const int* in_sizes, int n_in,
                              void* d_out, int out_size, void* d_ws, size_t ws_size,
                              hipStream_t stream) {
    const float*     x    = (const float*)d_in[0];
    const float*     W1   = (const float*)d_in[1];
    const float*     b1   = (const float*)d_in[2];
    const float*     W2   = (const float*)d_in[3];
    const float*     b2   = (const float*)d_in[4];
    const long long* eidx = (const long long*)d_in[5];
    const long long* rows = eidx;            // edge_index[0, :]
    const long long* cols = eidx + NEDGES;   // edge_index[1, :]
    float* out = (float*)d_out;

    // Workspace layout: dinv[N] | xw[N*128] (reused for xw2) | agg[N*128]
    //                   | W1 hi/lo packed | W2 hi/lo packed
    char* ws = (char*)d_ws;
    size_t off = 0;
    auto alloc = [&](size_t bytes) {
        char* p = ws + off;
        off = (off + bytes + 255) & ~(size_t)255;
        return p;
    };
    float*    dinv  = (float*)   alloc((size_t)NNODES * 4);
    float*    xw    = (float*)   alloc((size_t)NNODES * HIDC * 4);
    float*    agg   = (float*)   alloc((size_t)NNODES * HIDC * 4);
    unsigned* w1hiT = (unsigned*)alloc((size_t)HIDC * (INC / 2) * 4);   // [128][64]
    unsigned* w1loT = (unsigned*)alloc((size_t)HIDC * (INC / 2) * 4);
    unsigned* w2hiT = (unsigned*)alloc((size_t)EMBC * (HIDC / 2) * 4);  // [64][64]
    unsigned* w2loT = (unsigned*)alloc((size_t)EMBC * (HIDC / 2) * 4);

    const int TB = 256;
    // 0) pre-split weights to pair-packed transposed bf16 hi/lo
    k_splitW<<<(HIDC * (INC / 2) + TB - 1) / TB, TB, 0, stream>>>(W1, INC, HIDC, w1hiT, w1loT);
    k_splitW<<<(EMBC * (HIDC / 2) + TB - 1) / TB, TB, 0, stream>>>(W2, HIDC, EMBC, w2hiT, w2loT);

    // 1) degree with self-loops -> dinv
    k_deg_init <<<(NNODES + TB - 1) / TB, TB, 0, stream>>>(dinv);
    k_deg_count<<<(NEDGES + TB - 1) / TB, TB, 0, stream>>>(cols, dinv);
    k_dinv     <<<(NNODES + TB - 1) / TB, TB, 0, stream>>>(dinv);

    // 2) layer 1: xw1 = x @ W1 ; agg1 = scatter + self-loop
    k_gemm1<<<NNODES / 16, TB, 0, stream>>>(x, w1hiT, w1loT, xw);
    k_selfinit1<<<((size_t)NNODES * HIDC + TB - 1) / TB, TB, 0, stream>>>(xw, dinv, agg);
    k_edge1<<<(size_t)NEDGES * 32 / TB, TB, 0, stream>>>(rows, cols, dinv, xw, agg);

    // 3) layer 2: xw2 = relu(agg1 + b1) @ W2 ; out = scatter + self-loop + b2
    k_gemm2<<<NNODES / 32, TB, 0, stream>>>(agg, b1, w2hiT, w2loT, xw);
    k_selfinit2<<<((size_t)NNODES * EMBC + TB - 1) / TB, TB, 0, stream>>>(xw, dinv, b2, out);
    k_edge2<<<(size_t)NEDGES * 32 / TB, TB, 0, stream>>>(rows, cols, dinv, xw, out);
}
